// PerformerEncoderLayer_4277787427637
// MI455X (gfx1250) — matrix operand order repaired and checked by reading, not compiled, not run on hardware
//
#include <hip/hip_runtime.h>
#include <hip/hip_bf16.h>

// ---------------------------------------------------------------------------
// Performer encoder layer for MI455X (gfx1250), bf16 WMMA everywhere.
// S=4096 B=4 D=1024 H=16 DK=64 F=4096, tokens NT = S*B = 16384.
// ---------------------------------------------------------------------------

#define DEV __device__ __forceinline__

static constexpr int S_  = 4096;
static constexpr int B_  = 4;
static constexpr int D_  = 1024;
static constexpr int H_  = 16;
static constexpr int DK_ = 64;
static constexpr int F_  = 4096;
static constexpr int NT_ = S_ * B_;   // 16384 tokens

typedef __attribute__((ext_vector_type(16))) __bf16          v16bf;
typedef __attribute__((ext_vector_type(8)))  float           v8f;
typedef __attribute__((ext_vector_type(8)))  unsigned short  u16x8;
typedef __attribute__((ext_vector_type(16))) unsigned short  u16x16;

union Frag16 { u16x16 u; unsigned short s[16]; };

DEV unsigned short f2bf(float f) {       // round-to-nearest-even f32 -> bf16
  unsigned int u = __builtin_bit_cast(unsigned int, f);
  u += 0x7fffu + ((u >> 16) & 1u);
  return (unsigned short)(u >> 16);
}

DEV v8f wmma_bf16(v16bf a, v16bf b, v8f c) {
  // v_wmma_f32_16x16x32_bf16  (neg_a, A, neg_b, B, c_mod, C, reuse_a, reuse_b)
  return __builtin_amdgcn_wmma_f32_16x16x32_bf16(false, a, false, b,
                                                 (short)0, c, false, false);
}

// A fragment (16x32 bf16, M x K): lane<16 holds K 0..7 & 16..23 of row (lane&15),
// lane>=16 holds K 8..15 & 24..31 (ISA 7.12.2 16-bit A layout).
DEV v16bf load_a_frag(const unsigned short* tile, size_t ld, int lane) {
  const int row = lane & 15;
  const int ko  = (lane >> 4) << 3;              // 0 or 8
  const unsigned short* p = tile + (size_t)row * ld + ko;
  u16x8 lo = *(const u16x8*)(p);
  u16x8 hi = *(const u16x8*)(p + 16);
  Frag16 f;
#pragma unroll
  for (int i = 0; i < 8; ++i) { f.s[i] = lo[i]; f.s[8 + i] = hi[i]; }
  return __builtin_bit_cast(v16bf, f.u);
}

// B fragment (32x16 bf16, K x N): lane holds column (lane&15), lane<16 holds
// K 0..7 & 16..23, lane>=16 holds K 8..15 & 24..31 (matches A). Our B columns
// are rows of W (N x K row-major), so these are two 16-byte loads.
DEV v16bf load_b_frag(const unsigned short* wrow0, size_t ld, int lane) {
  const int col = lane & 15;
  const int ko  = (lane >> 4) << 3;              // 0 or 8
  const unsigned short* p = wrow0 + (size_t)col * ld + ko;
  u16x8 lo = *(const u16x8*)(p);
  u16x8 hi = *(const u16x8*)(p + 16);
  Frag16 f;
#pragma unroll
  for (int i = 0; i < 8; ++i) { f.s[i] = lo[i]; f.s[8 + i] = hi[i]; }
  return __builtin_bit_cast(v16bf, f.u);
}

// ---------------------------------------------------------------------------
// f32 -> bf16 conversion (weights + x)
// ---------------------------------------------------------------------------
__global__ __launch_bounds__(256) void cvt_kernel(const float* __restrict__ in,
                                                  unsigned short* __restrict__ out,
                                                  int n) {
  for (int i = blockIdx.x * blockDim.x + threadIdx.x; i < n;
       i += gridDim.x * blockDim.x)
    out[i] = f2bf(in[i]);
}

// ---------------------------------------------------------------------------
// Generic bf16 GEMM  C[M,N] = A[M,K] * W[N,K]^T (+bias) with epilogues.
// block = 256 threads = 8 waves; block tile 128x128; wave tile 32x64.
// EPI: 0 = elu+1 -> bf16 (q)        1 = elu+1, mask -> bf16 (k)
//      2 = bias -> bf16 (v)         3 = relu -> bf16 (ffn1)
//      4 = + residual -> f32 (attn-out / ffn2)
// ---------------------------------------------------------------------------
template <int EPI>
__global__ __launch_bounds__(256) void gemm_bf16_kernel(
    const unsigned short* __restrict__ A, const unsigned short* __restrict__ W,
    const float* __restrict__ bias, int M, int N, int K,
    const float* __restrict__ resid, unsigned short* __restrict__ outB,
    float* __restrict__ outF, const unsigned char* __restrict__ mask) {
  const int wave = threadIdx.x >> 5;
  const int lane = threadIdx.x & 31;
  const int m0 = blockIdx.y * 128 + (wave >> 1) * 32;
  const int n0 = blockIdx.x * 128 + (wave & 1) * 64;

  const unsigned short* a0p = A + (size_t)m0 * K;
  const unsigned short* a1p = A + (size_t)(m0 + 16) * K;

  v8f acc[2][4] = {};
  for (int k0 = 0; k0 < K; k0 += 32) {
    if (k0 + 32 < K)  // global_prefetch_b8 on upcoming weight tile
      __builtin_prefetch(W + (size_t)n0 * K + k0 + 32, 0, 1);
    v16bf a0 = load_a_frag(a0p + k0, K, lane);
    v16bf a1 = load_a_frag(a1p + k0, K, lane);
#pragma unroll
    for (int j = 0; j < 4; ++j) {
      v16bf bb = load_b_frag(W + (size_t)(n0 + 16 * j) * K + k0, K, lane);
      acc[0][j] = wmma_bf16(a0, bb, acc[0][j]);
      acc[1][j] = wmma_bf16(a1, bb, acc[1][j]);
    }
  }

  // C/D layout: VGPR r of lane -> row (lane>>4)*8 + r, col (lane&15).
  const int crow = (lane >> 4) * 8;
  const int ccol = lane & 15;
#pragma unroll
  for (int i = 0; i < 2; ++i) {
#pragma unroll
    for (int j = 0; j < 4; ++j) {
      const int n  = n0 + 16 * j + ccol;
      const float bv = bias[n];
#pragma unroll
      for (int r = 0; r < 8; ++r) {
        const int m = m0 + 16 * i + crow + r;
        float y = acc[i][j][r] + bv;
        const size_t idx = (size_t)m * N + n;
        if (EPI == 0 || EPI == 1) {
          y = (y > 0.f) ? (y + 1.f) : __expf(y);   // elu(y)+1
          if (EPI == 1) {
            const int bb_ = m & (B_ - 1);          // token = s*B + b
            const int ss  = m >> 2;
            if (mask[bb_ * S_ + ss]) y = 0.f;
          }
          outB[idx] = f2bf(y);
        } else if (EPI == 2) {
          outB[idx] = f2bf(y);
        } else if (EPI == 3) {
          outB[idx] = f2bf(fmaxf(y, 0.f));
        } else {
          outF[idx] = y + resid[idx];
        }
      }
    }
  }
}

// ---------------------------------------------------------------------------
// kv[b,h,d,e] = sum_s k[s,b,h,d] * v[s,b,h,e]   (K = S = 4096, strided axis)
// One block per (b,h); stage 32x64 k/v tiles in LDS, gather transposed A
// fragments from LDS, WMMA-accumulate 64x64 f32. Store e-major bf16 so the
// result feeds qkv_kernel as a B operand.
// ---------------------------------------------------------------------------
__global__ __launch_bounds__(256) void kv_kernel(
    const unsigned short* __restrict__ Kb, const unsigned short* __restrict__ Vb,
    unsigned short* __restrict__ KVc) {
  const int bh = blockIdx.x;
  const int b  = bh / H_;
  const int h  = bh % H_;
  __shared__ __align__(16) unsigned short kl[32][64];
  __shared__ __align__(16) unsigned short vl[32][64];

  const int tid  = threadIdx.x;
  const int lane = tid & 31;
  const int wave = tid >> 5;
  const int tile_m = wave >> 1;          // d tile 0..3
  const int tn0    = (wave & 1) * 2;     // e tiles tn0, tn0+1

  v8f acc[2] = {};
  const int ls = tid >> 3;               // 0..31 : s row
  const int lc = (tid & 7) * 8;          // col start (8 ushorts)

  for (int s0 = 0; s0 < S_; s0 += 32) {
    const size_t tok = (size_t)(s0 + ls) * B_ + b;
    u16x8 kv8 = *(const u16x8*)(Kb + tok * D_ + h * DK_ + lc);
    u16x8 vv8 = *(const u16x8*)(Vb + tok * D_ + h * DK_ + lc);
    __syncthreads();
    *(u16x8*)&kl[ls][lc] = kv8;
    *(u16x8*)&vl[ls][lc] = vv8;
    __syncthreads();

    // A = k^T : row = d, K = s (transposed gather from LDS)
    Frag16 af;
    const int d  = tile_m * 16 + (lane & 15);
    const int sb = (lane >> 4) * 8;
#pragma unroll
    for (int i = 0; i < 8; ++i) {
      af.s[i]     = kl[sb + i][d];
      af.s[8 + i] = kl[16 + sb + i][d];
    }
    v16bf a = __builtin_bit_cast(v16bf, af.u);
#pragma unroll
    for (int j = 0; j < 2; ++j) {
      Frag16 bf;
      const int e  = (tn0 + j) * 16 + (lane & 15);
      const int sv = (lane >> 4) * 8;
#pragma unroll
      for (int i = 0; i < 16; ++i) bf.s[i] = vl[(i < 8) ? (sv + i) : (16 + sv + i - 8)][e];
      acc[j] = wmma_bf16(a, __builtin_bit_cast(v16bf, bf.u), acc[j]);
    }
  }

  const int crow = (lane >> 4) * 8;
  const int ccol = lane & 15;
  unsigned short* out = KVc + (size_t)bh * DK_ * DK_;
#pragma unroll
  for (int j = 0; j < 2; ++j) {
    const int e = (tn0 + j) * 16 + ccol;
#pragma unroll
    for (int r = 0; r < 8; ++r) {
      const int d = tile_m * 16 + crow + r;
      out[e * DK_ + d] = f2bf(acc[j][r]);   // e-major: column d contiguous
    }
  }
}

// ---------------------------------------------------------------------------
// qkv[s,b,h,e] = sum_d q[s,b,h,d] * kv[b,h,d,e]   (per-head GEMM, K = 64)
// grid = (S/256, B*H); A rows are strided by B*D (fixed b), still b128 loads.
// ---------------------------------------------------------------------------
__global__ __launch_bounds__(256) void qkv_kernel(
    const unsigned short* __restrict__ Qb, const unsigned short* __restrict__ KVc,
    unsigned short* __restrict__ Ob) {
  const int bh = blockIdx.y;
  const int b  = bh / H_;
  const int h  = bh % H_;
  const int wave = threadIdx.x >> 5;
  const int lane = threadIdx.x & 31;
  const int s0 = blockIdx.x * 256 + wave * 32;
  const unsigned short* kvp = KVc + (size_t)bh * DK_ * DK_;

  v8f acc[2][4] = {};
  for (int k0 = 0; k0 < DK_; k0 += 32) {
    v16bf a0 = load_a_frag(Qb + ((size_t)s0 * B_ + b) * D_ + h * DK_ + k0,
                           (size_t)B_ * D_, lane);
    v16bf a1 = load_a_frag(Qb + ((size_t)(s0 + 16) * B_ + b) * D_ + h * DK_ + k0,
                           (size_t)B_ * D_, lane);
#pragma unroll
    for (int j = 0; j < 4; ++j) {
      v16bf bb = load_b_frag(kvp + (size_t)(16 * j) * DK_ + k0, DK_, lane);
      acc[0][j] = wmma_bf16(a0, bb, acc[0][j]);
      acc[1][j] = wmma_bf16(a1, bb, acc[1][j]);
    }
  }

  const int crow = (lane >> 4) * 8;
  const int ccol = lane & 15;
#pragma unroll
  for (int i = 0; i < 2; ++i) {
#pragma unroll
    for (int j = 0; j < 4; ++j) {
      const int e = 16 * j + ccol;
#pragma unroll
      for (int r = 0; r < 8; ++r) {
        const int s = s0 + 16 * i + crow + r;
        Ob[((size_t)s * B_ + b) * D_ + h * DK_ + e] = f2bf(acc[i][j][r]);
      }
    }
  }
}

// ---------------------------------------------------------------------------
// Row LayerNorm over D=1024; one block per token. Writes f32 and (optional)
// bf16 copy for the next GEMM.
// ---------------------------------------------------------------------------
__global__ __launch_bounds__(256) void ln_kernel(
    const float* __restrict__ in, const float* __restrict__ g,
    const float* __restrict__ be, float* __restrict__ outF,
    unsigned short* __restrict__ outB) {
  const int row = blockIdx.x;
  const int t   = threadIdx.x;
  const float* p = in + (size_t)row * D_;
  float v[4], s = 0.f, ss = 0.f;
#pragma unroll
  for (int i = 0; i < 4; ++i) {
    float x = p[t + 256 * i];
    v[i] = x; s += x; ss += x * x;
  }
  __shared__ float rs[256], rss[256];
  rs[t] = s; rss[t] = ss;
  __syncthreads();
  for (int off = 128; off > 0; off >>= 1) {
    if (t < off) { rs[t] += rs[t + off]; rss[t] += rss[t + off]; }
    __syncthreads();
  }
  const float mean = rs[0] * (1.f / D_);
  const float var  = rss[0] * (1.f / D_) - mean * mean;
  const float rstd = rsqrtf(var + 1e-5f);
  float* o = outF + (size_t)row * D_;
#pragma unroll
  for (int i = 0; i < 4; ++i) {
    const int c = t + 256 * i;
    const float y = (v[i] - mean) * rstd * g[c] + be[c];
    o[c] = y;
    if (outB) outB[(size_t)row * D_ + c] = f2bf(y);
  }
}

// ---------------------------------------------------------------------------
// Launch
// ---------------------------------------------------------------------------
extern "C" void kernel_launch(void* const* d_in, const int* in_sizes, int n_in,
                              void* d_out, int out_size, void* d_ws, size_t ws_size,
                              hipStream_t stream) {
  (void)in_sizes; (void)n_in; (void)out_size; (void)ws_size;
  const float* x  = (const float*)d_in[0];
  const unsigned char* mask = (const unsigned char*)d_in[1];
  const float* Wq = (const float*)d_in[2];  const float* bq = (const float*)d_in[3];
  const float* Wk = (const float*)d_in[4];  const float* bk = (const float*)d_in[5];
  const float* Wv = (const float*)d_in[6];  const float* bv = (const float*)d_in[7];
  const float* Wo = (const float*)d_in[8];  const float* bo = (const float*)d_in[9];
  const float* W1 = (const float*)d_in[10]; const float* b1 = (const float*)d_in[11];
  const float* W2 = (const float*)d_in[12]; const float* b2 = (const float*)d_in[13];
  const float* g1 = (const float*)d_in[14]; const float* be1 = (const float*)d_in[15];
  const float* g2 = (const float*)d_in[16]; const float* be2 = (const float*)d_in[17];

  // workspace layout (ushort counts); total ~345 MB
  const size_t DD = (size_t)D_ * D_;         // 1M
  const size_t DF = (size_t)D_ * F_;         // 4M
  const size_t ND = (size_t)NT_ * D_;        // 16M
  unsigned short* us  = (unsigned short*)d_ws;
  unsigned short* WQb = us;
  unsigned short* WKb = WQb + DD;
  unsigned short* WVb = WKb + DD;
  unsigned short* WOb = WVb + DD;
  unsigned short* W1b = WOb + DD;
  unsigned short* W2b = W1b + DF;
  unsigned short* XB   = W2b + DF;           // x bf16
  unsigned short* QB   = XB + ND;
  unsigned short* KB   = QB + ND;
  unsigned short* VB   = KB + ND;
  unsigned short* QKVB = VB + ND;
  unsigned short* KVC  = QKVB + ND;          // B*H*64*64
  unsigned short* X1B  = KVC + (size_t)B_ * H_ * DK_ * DK_;
  unsigned short* FF1  = XB;                 // alias dead XB..VB region (NT*F)
  float* R1  = (float*)(X1B + ND);           // pre-LN residual (reused twice)
  float* X1F = R1 + ND;

  const dim3 blk(256);

  // 1) bf16 conversions
  cvt_kernel<<<2048, blk, 0, stream>>>(Wq, WQb, (int)DD);
  cvt_kernel<<<2048, blk, 0, stream>>>(Wk, WKb, (int)DD);
  cvt_kernel<<<2048, blk, 0, stream>>>(Wv, WVb, (int)DD);
  cvt_kernel<<<2048, blk, 0, stream>>>(Wo, WOb, (int)DD);
  cvt_kernel<<<4096, blk, 0, stream>>>(W1, W1b, (int)DF);
  cvt_kernel<<<4096, blk, 0, stream>>>(W2, W2b, (int)DF);
  cvt_kernel<<<8192, blk, 0, stream>>>(x,  XB,  (int)ND);

  // 2) q,k,v projections (WMMA), feature map + mask fused in epilogue
  gemm_bf16_kernel<0><<<dim3(D_ / 128, NT_ / 128), blk, 0, stream>>>(
      XB, WQb, bq, NT_, D_, D_, nullptr, QB, nullptr, nullptr);
  gemm_bf16_kernel<1><<<dim3(D_ / 128, NT_ / 128), blk, 0, stream>>>(
      XB, WKb, bk, NT_, D_, D_, nullptr, KB, nullptr, mask);
  gemm_bf16_kernel<2><<<dim3(D_ / 128, NT_ / 128), blk, 0, stream>>>(
      XB, WVb, bv, NT_, D_, D_, nullptr, VB, nullptr, nullptr);

  // 3) kv = k^T v per (b,h); 4) qkv = q @ kv
  kv_kernel<<<B_ * H_, blk, 0, stream>>>(KB, VB, KVC);
  qkv_kernel<<<dim3(S_ / 256, B_ * H_), blk, 0, stream>>>(QB, KVC, QKVB);

  // 5) attn = qkv @ Wo^T + bo (+x residual), then LN1
  gemm_bf16_kernel<4><<<dim3(D_ / 128, NT_ / 128), blk, 0, stream>>>(
      QKVB, WOb, bo, NT_, D_, D_, x, nullptr, R1, nullptr);
  ln_kernel<<<NT_, blk, 0, stream>>>(R1, g1, be1, X1F, X1B);

  // 6) FFN: relu(x1 @ W1^T + b1) @ W2^T + b2 (+x1 residual), then LN2 -> out
  gemm_bf16_kernel<3><<<dim3(F_ / 128, NT_ / 128), blk, 0, stream>>>(
      X1B, W1b, b1, NT_, F_, D_, nullptr, FF1, nullptr, nullptr);
  gemm_bf16_kernel<4><<<dim3(D_ / 128, NT_ / 128), blk, 0, stream>>>(
      FF1, W2b, b2, NT_, D_, F_, X1F, nullptr, R1, nullptr);
  ln_kernel<<<NT_, blk, 0, stream>>>(R1, g2, be2, (float*)d_out, nullptr);
}
